// Model_12747462934681
// MI455X (gfx1250) — matrix operand order repaired and checked by reading, not compiled, not run on hardware
//
#include <hip/hip_runtime.h>
#include <hip/hip_bf16.h>

typedef __attribute__((ext_vector_type(16))) _Float16 v16h;
typedef __attribute__((ext_vector_type(8)))  float    v8f;

constexpr int Hd = 512;    // d_model
constexpr int Bn = 64;     // batch
constexpr int Le = 512;    // encoder length
constexpr int Sd = 48;     // decoder prediction steps (96-48)
constexpr int G4 = 2048;   // 4*Hd gate width

// ---------------- WMMA fragment helpers (wave32; contiguous k order chosen by this file) ----------------
// A 16x32 f16 (MxK): lanes 0-15 row M=lane, elems 0..15 -> K 0..15
//                    lanes 16-31 same rows,  elems 0..15 -> K 16..31
__device__ __forceinline__ v16h load_A16(const _Float16* __restrict__ A, int m0, int k0, int lda) {
  int lane = threadIdx.x & 31;
  int m  = m0 + (lane & 15);
  int kb = k0 + ((lane >> 4) << 4);
  const _Float16* p = A + (size_t)m * lda + kb;
  v16h a;
#pragma unroll
  for (int e = 0; e < 16; ++e) a[e] = p[e];
  return a;
}

// B 32x16 f16 (KxN) from weight W stored row-major (N rows x K cols): B[k][n] = W[n][k].
// lanes 0-15: N=lane, K 0..15 ; lanes 16-31: N=lane-16, K 16..31  (contiguous per lane)
__device__ __forceinline__ v16h load_BT16(const _Float16* __restrict__ W, int n0, int k0, int ldw) {
  int lane = threadIdx.x & 31;
  int n  = n0 + (lane & 15);
  int kb = k0 + ((lane >> 4) << 4);
  const _Float16* p = W + (size_t)n * ldw + kb;
  v16h b;
#pragma unroll
  for (int e = 0; e < 16; ++e) b[e] = p[e];
  return b;
}

__device__ __forceinline__ v8f wmma_f16(v16h a, v16h b, v8f c) {
  return __builtin_amdgcn_wmma_f32_16x16x32_f16(false, a, false, b, (short)0, c, false, false);
}

__device__ __forceinline__ float sigf(float x) { return 1.f / (1.f + __expf(-x)); }

// ---------------- generic GEMM: C(MxN) = A(MxK,f16) * W(NxK,f16)^T + bias ----------------
// block = 256 threads (8 waves), tile 64(M) x 128(N); wave -> 1 Mtile x 4 Ntiles.
// Software-pipelined: next k-step fragments are loaded before current WMMAs issue.
template <bool OUT_HALF>
__global__ void gemm_nt(const _Float16* __restrict__ A, int lda,
                        const _Float16* __restrict__ W, int ldw,
                        const float* __restrict__ bias,
                        void* __restrict__ Cout, int ldc, int K) {
  int wave = threadIdx.x >> 5;
  int m0 = blockIdx.y * 64 + (wave >> 1) * 16;
  int n0 = blockIdx.x * 128 + (wave & 1) * 64;
  v8f zero = {0.f, 0.f, 0.f, 0.f, 0.f, 0.f, 0.f, 0.f};
  v8f acc[4];
#pragma unroll
  for (int j = 0; j < 4; ++j) acc[j] = zero;

  v16h a = load_A16(A, m0, 0, lda);
  v16h b[4];
#pragma unroll
  for (int j = 0; j < 4; ++j) b[j] = load_BT16(W, n0 + j * 16, 0, ldw);

  for (int k0 = 0; k0 < K; k0 += 32) {
    int kn = (k0 + 32 < K) ? (k0 + 32) : k0;     // clamp: keep loads in-bounds
    v16h an = load_A16(A, m0, kn, lda);
    v16h bn[4];
#pragma unroll
    for (int j = 0; j < 4; ++j) bn[j] = load_BT16(W, n0 + j * 16, kn, ldw);
#pragma unroll
    for (int j = 0; j < 4; ++j) acc[j] = wmma_f16(a, b[j], acc[j]);
    a = an;
#pragma unroll
    for (int j = 0; j < 4; ++j) b[j] = bn[j];
  }

  int lane = threadIdx.x & 31;
  int nl = lane & 15;
  int mb = m0 + ((lane >> 4) << 3);
#pragma unroll
  for (int j = 0; j < 4; ++j) {
    int n = n0 + j * 16 + nl;
    float bs = bias ? bias[n] : 0.f;
#pragma unroll
    for (int r = 0; r < 8; ++r) {
      float val = acc[j][r] + bs;
      if constexpr (OUT_HALF) ((_Float16*)Cout)[(size_t)(mb + r) * ldc + n] = (_Float16)val;
      else                    ((float*)Cout)[(size_t)(mb + r) * ldc + n] = val;
    }
  }
}

// ---------------- fused LSTM step: g = xg + h_prev @ Whh^T ; gate math ----------------
// grid = 32 blocks; block wg owns hidden cols [wg*16, wg*16+16) across all 4 gates
// -> 16 C tiles per block, 2 per wave (wave = mt(4) x qg(2); tiles q=2*qg, 2*qg+1).
// Software-pipelined fragment loads; gate epilogue via a 16KB LDS tile.
__global__ void lstm_step(const _Float16* __restrict__ h_prev, int ld_h,
                          const _Float16* __restrict__ Whh,          // (2048 x 512) f16
                          const _Float16* __restrict__ xg, int xg_bstride,
                          float* __restrict__ c_buf,
                          float* __restrict__ h_out_f, int hf_bstride,
                          _Float16* __restrict__ h_out_h, int hh_bstride,
                          int has_hprev) {
  __shared__ float gls[64 * 64];    // 64 batch rows x (4 gates * 16 local cols)
  const int wg   = blockIdx.x;      // 0..31
  const int wave = threadIdx.x >> 5;
  const int mt   = wave & 3;
  const int qg   = wave >> 2;       // 0: gates {i,f}, 1: gates {g,o}
  const int m0   = mt * 16;
  v8f zero = {0.f, 0.f, 0.f, 0.f, 0.f, 0.f, 0.f, 0.f};
  v8f acc0 = zero, acc1 = zero;

  if (has_hprev) {
    const int n0 = (2 * qg + 0) * Hd + wg * 16;
    const int n1 = (2 * qg + 1) * Hd + wg * 16;
    v16h a  = load_A16(h_prev, m0, 0, ld_h);
    v16h b0 = load_BT16(Whh, n0, 0, Hd);
    v16h b1 = load_BT16(Whh, n1, 0, Hd);
    for (int k0 = 0; k0 < Hd; k0 += 32) {
      int kn = (k0 + 32 < Hd) ? (k0 + 32) : k0;
      v16h an  = load_A16(h_prev, m0, kn, ld_h);
      v16h b0n = load_BT16(Whh, n0, kn, Hd);
      v16h b1n = load_BT16(Whh, n1, kn, Hd);
      acc0 = wmma_f16(a, b0, acc0);
      acc1 = wmma_f16(a, b1, acc1);
      a = an; b0 = b0n; b1 = b1n;
    }
  }

  int lane = threadIdx.x & 31;
  int nl = lane & 15;
  int rb = (lane >> 4) << 3;
#pragma unroll
  for (int r = 0; r < 8; ++r) {
    gls[(m0 + rb + r) * 64 + (2 * qg + 0) * 16 + nl] = acc0[r];
    gls[(m0 + rb + r) * 64 + (2 * qg + 1) * 16 + nl] = acc1[r];
  }
  __syncthreads();

#pragma unroll
  for (int e = 0; e < 4; ++e) {             // 1024 (b, j) elements / 256 threads
    int idx = e * 256 + threadIdx.x;
    int b = idx >> 4;
    int j = idx & 15;
    int hg = wg * 16 + j;
    const _Float16* xr = xg + (size_t)b * xg_bstride;
    float gi = (float)xr[0 * Hd + hg] + gls[b * 64 + 0 * 16 + j];
    float gf = (float)xr[1 * Hd + hg] + gls[b * 64 + 1 * 16 + j];
    float gg = (float)xr[2 * Hd + hg] + gls[b * 64 + 2 * 16 + j];
    float go = (float)xr[3 * Hd + hg] + gls[b * 64 + 3 * 16 + j];
    float cold = has_hprev ? c_buf[b * Hd + hg] : 0.f;
    float c = sigf(gf) * cold + sigf(gi) * tanhf(gg);
    float h = sigf(go) * tanhf(c);
    c_buf[b * Hd + hg] = c;
    if (h_out_f) h_out_f[(size_t)b * hf_bstride + hg] = h;
    h_out_h[(size_t)b * hh_bstride + hg] = (_Float16)h;
  }
}

// ---------------- conv1d(7->512,k=3,wrap) embedding + sinusoidal PE, f16 out ----------------
__global__ void embed_kernel(const float* __restrict__ x, const float* __restrict__ w,
                             _Float16* __restrict__ out, int Lx, int t0, int Tout) {
  int idx = blockIdx.x * blockDim.x + threadIdx.x;
  int total = Bn * Tout * Hd;
  if (idx >= total) return;
  int o = idx % Hd;
  int s = (idx / Hd) % Tout;
  int b = idx / (Hd * Tout);
  int t = t0 + s;
  float accv = 0.f;
#pragma unroll
  for (int r = 0; r < 3; ++r) {
    int tt = t - 1 + r; tt += Lx; tt %= Lx;            // 'wrap' padding
    const float* xp = x + (size_t)(b * Lx + tt) * 7;
    const float* wp = w + o * 21 + r;                  // w[o][i][r]
#pragma unroll
    for (int i = 0; i < 7; ++i) accv += xp[i] * wp[i * 3];
  }
  float j2  = (float)(o & ~1);
  float div = __expf(-j2 * (logf(10000.f) / (float)Hd));
  float ang = (float)t * div;
  accv += (o & 1) ? __cosf(ang) : __sinf(ang);
  out[idx] = (_Float16)accv;
}

// ---------------- attention: scores[b,l] = sum_k tanh(enc_part + s1)*v ----------------
__global__ void attn_scores(const float* __restrict__ enc_part, const float* __restrict__ s1,
                            const float* __restrict__ v, float* __restrict__ scores) {
  int wave = threadIdx.x >> 5;
  int lane = threadIdx.x & 31;
  int gid = blockIdx.x * 8 + wave;
  int b = gid >> 9;
  int l = gid & 511;
  const float* ep  = enc_part + (size_t)(b * Le + l) * Hd;
  const float* s1b = s1 + b * Hd;
  float acc = 0.f;
  for (int k = lane; k < Hd; k += 32) acc += tanhf(ep[k] + s1b[k]) * v[k];
#pragma unroll
  for (int off = 16; off > 0; off >>= 1) acc += __shfl_down(acc, off, 32);
  if (lane == 0) scores[b * Le + l] = acc;
}

// ---------------- softmax over L then ctx = sum_l aw[l]*enc_out[b,l,:] -> concat[:,512:] ----------------
__global__ void softmax_ctx(const float* __restrict__ scores, const float* __restrict__ enc_out_f,
                            _Float16* __restrict__ concat) {
  __shared__ float aw[512];
  __shared__ float red[256];
  int b = blockIdx.x, tid = threadIdx.x;
  float sa = scores[b * Le + tid];
  float sb = scores[b * Le + 256 + tid];
  red[tid] = fmaxf(sa, sb);
  __syncthreads();
  for (int off = 128; off > 0; off >>= 1) { if (tid < off) red[tid] = fmaxf(red[tid], red[tid + off]); __syncthreads(); }
  float m = red[0];
  __syncthreads();
  float ea = __expf(sa - m), eb = __expf(sb - m);
  aw[tid] = ea; aw[tid + 256] = eb;
  red[tid] = ea + eb;
  __syncthreads();
  for (int off = 128; off > 0; off >>= 1) { if (tid < off) red[tid] += red[tid + off]; __syncthreads(); }
  float inv = 1.f / red[0];
  float a0 = 0.f, a1 = 0.f;
  const float* eo = enc_out_f + (size_t)b * Le * Hd;
  for (int l = 0; l < Le; ++l) {
    float wl = aw[l];
    a0 += wl * eo[(size_t)l * Hd + tid];
    a1 += wl * eo[(size_t)l * Hd + 256 + tid];
  }
  concat[b * 1024 + Hd + tid]       = (_Float16)(a0 * inv);
  concat[b * 1024 + Hd + 256 + tid] = (_Float16)(a1 * inv);
}

// ---------------- small helpers ----------------
__global__ void cvt_f32_f16(const float* __restrict__ src, _Float16* __restrict__ dst, int n) {
  int i = blockIdx.x * blockDim.x + threadIdx.x;
  if (i < n) dst[i] = (_Float16)src[i];
}
__global__ void cvt_strided(const float* __restrict__ src, _Float16* __restrict__ dst,
                            int rows, int cols, int src_ld, int src_off) {
  int i = blockIdx.x * blockDim.x + threadIdx.x;
  if (i >= rows * cols) return;
  int r = i / cols, c = i % cols;
  dst[i] = (_Float16)src[(size_t)r * src_ld + src_off + c];
}
__global__ void init_dec(const _Float16* __restrict__ enc_out_h, _Float16* __restrict__ concat0) {
  int i = blockIdx.x * blockDim.x + threadIdx.x;
  if (i >= Bn * Hd) return;
  int b = i >> 9, h = i & 511;
  concat0[b * 1024 + h] = enc_out_h[(size_t)(b * Le + (Le - 1)) * Hd + h];
}
__global__ void proj_kernel(const float* __restrict__ outs, const float* __restrict__ pw,
                            const float* __restrict__ pb, float* __restrict__ out) {
  int i = blockIdx.x * blockDim.x + threadIdx.x;
  if (i >= Bn * Sd * 7) return;
  int p = i % 7;
  int s = (i / 7) % Sd;
  int b = i / (7 * Sd);
  const float* row = outs + (size_t)(b * Sd + s) * Hd;
  float acc = pb[p];
  for (int h = 0; h < Hd; ++h) acc += row[h] * pw[p * Hd + h];
  out[i] = acc;
}

extern "C" void kernel_launch(void* const* d_in, const int* in_sizes, int n_in,
                              void* d_out, int out_size, void* d_ws, size_t ws_size,
                              hipStream_t stream) {
  (void)in_sizes; (void)n_in; (void)out_size; (void)ws_size;
  const float* x_enc     = (const float*)d_in[0];
  const float* x_dec     = (const float*)d_in[1];
  const float* emb_enc_w = (const float*)d_in[2];
  const float* emb_dec_w = (const float*)d_in[3];
  const float* enc_Wih   = (const float*)d_in[4];
  const float* enc_Whh   = (const float*)d_in[5];
  const float* enc_b     = (const float*)d_in[6];
  const float* dec_Wih   = (const float*)d_in[7];
  const float* dec_Whh   = (const float*)d_in[8];
  const float* dec_b     = (const float*)d_in[9];
  const float* attn_W    = (const float*)d_in[10];
  const float* attn_b    = (const float*)d_in[11];
  const float* vvec      = (const float*)d_in[12];
  const float* fc_W      = (const float*)d_in[13];
  const float* fc_b      = (const float*)d_in[14];
  const float* proj_W    = (const float*)d_in[15];
  const float* proj_b    = (const float*)d_in[16];

  char* base = (char*)d_ws;
  size_t off = 0;
  auto alloc = [&](size_t bytes) -> char* {
    size_t o = off;
    off = (off + bytes + 255) & ~(size_t)255;
    return base + o;
  };
  _Float16* Wih_h   = (_Float16*)alloc((size_t)G4 * Hd * 2);
  _Float16* Whh_h   = (_Float16*)alloc((size_t)G4 * Hd * 2);
  _Float16* dWih_h  = (_Float16*)alloc((size_t)G4 * Hd * 2);
  _Float16* dWhh_h  = (_Float16*)alloc((size_t)G4 * Hd * 2);
  _Float16* fcW_h   = (_Float16*)alloc((size_t)Hd * 1024 * 2);
  _Float16* Wh_h    = (_Float16*)alloc((size_t)Hd * Hd * 2);
  _Float16* We_h    = (_Float16*)alloc((size_t)Hd * Hd * 2);
  _Float16* enc_in_h  = (_Float16*)alloc((size_t)Bn * Le * Hd * 2);
  _Float16* dec_in_h  = (_Float16*)alloc((size_t)Bn * Sd * Hd * 2);
  _Float16* xg_enc_h  = (_Float16*)alloc((size_t)Bn * Le * G4 * 2);
  _Float16* xg_dec_h  = (_Float16*)alloc((size_t)Bn * Sd * G4 * 2);
  float*    enc_out_f = (float*)alloc((size_t)Bn * Le * Hd * 4);
  _Float16* enc_out_h = (_Float16*)alloc((size_t)Bn * Le * Hd * 2);
  float*    enc_part_f = (float*)alloc((size_t)Bn * Le * Hd * 4);
  float*    c_buf   = (float*)alloc((size_t)Bn * Hd * 4);
  _Float16* concat0 = (_Float16*)alloc((size_t)Bn * 1024 * 2);
  _Float16* concat1 = (_Float16*)alloc((size_t)Bn * 1024 * 2);
  float*    s1_f    = (float*)alloc((size_t)Bn * Hd * 4);
  float*    scores_f = (float*)alloc((size_t)Bn * Le * 4);
  float*    outs_f  = (float*)alloc((size_t)Bn * Sd * Hd * 4);
  _Float16* concat[2] = {concat0, concat1};

  // weights -> f16
  int n = G4 * Hd;
  cvt_f32_f16<<<(n + 255) / 256, 256, 0, stream>>>(enc_Wih, Wih_h, n);
  cvt_f32_f16<<<(n + 255) / 256, 256, 0, stream>>>(enc_Whh, Whh_h, n);
  cvt_f32_f16<<<(n + 255) / 256, 256, 0, stream>>>(dec_Wih, dWih_h, n);
  cvt_f32_f16<<<(n + 255) / 256, 256, 0, stream>>>(dec_Whh, dWhh_h, n);
  n = Hd * 1024;
  cvt_f32_f16<<<(n + 255) / 256, 256, 0, stream>>>(fc_W, fcW_h, n);
  cvt_strided<<<(Hd * Hd + 255) / 256, 256, 0, stream>>>(attn_W, Wh_h, Hd, Hd, 2 * Hd, 0);
  cvt_strided<<<(Hd * Hd + 255) / 256, 256, 0, stream>>>(attn_W, We_h, Hd, Hd, 2 * Hd, Hd);

  // embeddings (decoder: only positions 48..95 are consumed)
  embed_kernel<<<(Bn * Le * Hd + 255) / 256, 256, 0, stream>>>(x_enc, emb_enc_w, enc_in_h, Le, 0, Le);
  embed_kernel<<<(Bn * Sd * Hd + 255) / 256, 256, 0, stream>>>(x_dec, emb_dec_w, dec_in_h, 144, 48, Sd);

  // time-parallel input GEMMs (bias folded in)
  gemm_nt<true><<<dim3(G4 / 128, (Bn * Le) / 64), 256, 0, stream>>>(enc_in_h, Hd, Wih_h, Hd, enc_b, xg_enc_h, G4, Hd);
  gemm_nt<true><<<dim3(G4 / 128, (Bn * Sd) / 64), 256, 0, stream>>>(dec_in_h, Hd, dWih_h, Hd, dec_b, xg_dec_h, G4, Hd);

  // encoder recurrence: one fused GEMM+gate kernel per step (graph serializes)
  for (int t = 0; t < Le; ++t) {
    lstm_step<<<32, 256, 0, stream>>>(
        enc_out_h + (size_t)(t > 0 ? t - 1 : 0) * Hd, Le * Hd,
        Whh_h,
        xg_enc_h + (size_t)t * G4, Le * G4,
        c_buf,
        enc_out_f + (size_t)t * Hd, Le * Hd,
        enc_out_h + (size_t)t * Hd, Le * Hd,
        t > 0 ? 1 : 0);
  }

  // enc_part = enc_out @ We^T + attn_b   (fp32 out, feeds tanh)
  gemm_nt<false><<<dim3(Hd / 128, (Bn * Le) / 64), 256, 0, stream>>>(enc_out_h, Hd, We_h, Hd, attn_b, enc_part_f, Hd, Hd);

  // decoder init: h0 = last encoder h (c0 already in c_buf)
  init_dec<<<(Bn * Hd + 255) / 256, 256, 0, stream>>>(enc_out_h, concat[0]);

  for (int s = 0; s < Sd; ++s) {
    _Float16* cprev = concat[s & 1];
    _Float16* cnext = concat[(s + 1) & 1];
    lstm_step<<<32, 256, 0, stream>>>(cprev, 1024, dWhh_h,
        xg_dec_h + (size_t)s * G4, Sd * G4,
        c_buf, nullptr, 0, cnext, 1024, 1);
    // s1 = h @ Wh^T
    gemm_nt<false><<<dim3(Hd / 128, 1), 256, 0, stream>>>(cnext, 1024, Wh_h, Hd, nullptr, s1_f, Hd, Hd);
    attn_scores<<<(Bn * Le) / 8, 256, 0, stream>>>(enc_part_f, s1_f, vvec, scores_f);
    softmax_ctx<<<Bn, 256, 0, stream>>>(scores_f, enc_out_f, cnext);
    // out_s = [h, ctx] @ fc_W^T + fc_b  -> outs[b][s][:]
    gemm_nt<false><<<dim3(Hd / 128, 1), 256, 0, stream>>>(cnext, 1024, fcW_h, 1024, fc_b,
                                                          outs_f + (size_t)s * Hd, Sd * Hd, 1024);
  }

  proj_kernel<<<(Bn * Sd * 7 + 255) / 256, 256, 0, stream>>>(outs_f, proj_W, proj_b, (float*)d_out);
}
